// MixtralMoE_50835232916127
// MI455X (gfx1250) — matrix operand order repaired and checked by reading, not compiled, not run on hardware
//
#include <hip/hip_runtime.h>

// Problem constants (from reference)
#define T 16384
#define H 1024
#define FF 2048
#define E 8

// MoE tile config
#define MT 32      // tokens per block
#define FC 64      // F chunk width
#define NWAVE 8    // waves per block (256 threads, wave32)

typedef __attribute__((ext_vector_type(16))) __bf16 v16bf;
typedef __attribute__((ext_vector_type(8)))  float  v8f;

union Frag {
    v16bf v;
    unsigned u[8];
};

// ---------------------------------------------------------------------------
// Kernel 0: zero output + expert counters
// ---------------------------------------------------------------------------
__global__ void zero_init_kernel(float* __restrict__ out, int* __restrict__ counts) {
    long i4 = ((long)blockIdx.x * blockDim.x + threadIdx.x) * 4;
    if (i4 < (long)T * H) {
        *(float4*)(out + i4) = make_float4(0.f, 0.f, 0.f, 0.f);
    }
    if (blockIdx.x == 0 && threadIdx.x < E) counts[threadIdx.x] = 0;
}

// ---------------------------------------------------------------------------
// Kernel 1: router. One wave (32 lanes) per token.
// logits = x . gate_w  -> softmax -> top2 -> renorm -> gates + expert lists
// ---------------------------------------------------------------------------
__global__ __launch_bounds__(256) void router_kernel(
        const float* __restrict__ hs, const float* __restrict__ gw,
        float* __restrict__ gates, int* __restrict__ counts, int* __restrict__ lists) {
    int wave = threadIdx.x >> 5;
    int lane = threadIdx.x & 31;
    int t = blockIdx.x * NWAVE + wave;

    float acc[E];
#pragma unroll
    for (int e = 0; e < E; ++e) acc[e] = 0.f;

    const float* x = hs + (size_t)t * H;
    for (int h = lane; h < H; h += 32) {
        float xv = x[h];
        const float* g = gw + (size_t)h * E;
#pragma unroll
        for (int e = 0; e < E; ++e) acc[e] += xv * g[e];
    }
#pragma unroll
    for (int e = 0; e < E; ++e) {
#pragma unroll
        for (int off = 16; off > 0; off >>= 1)
            acc[e] += __shfl_xor(acc[e], off, 32);
    }

    if (lane == 0) {
        float m = acc[0];
#pragma unroll
        for (int e = 1; e < E; ++e) m = fmaxf(m, acc[e]);
        float p[E];
#pragma unroll
        for (int e = 0; e < E; ++e) p[e] = __expf(acc[e] - m);
        // top-2 (ties -> lowest index, matching jax top_k)
        int i0 = 0;
#pragma unroll
        for (int e = 1; e < E; ++e) if (p[e] > p[i0]) i0 = e;
        int i1 = (i0 == 0) ? 1 : 0;
#pragma unroll
        for (int e = 0; e < E; ++e)
            if (e != i0 && p[e] > p[i1]) i1 = e;
        float w0 = p[i0], w1 = p[i1];
        float inv = 1.0f / (w0 + w1);
        w0 *= inv; w1 *= inv;
#pragma unroll
        for (int e = 0; e < E; ++e)
            gates[(size_t)t * E + e] = (e == i0) ? w0 : ((e == i1) ? w1 : 0.f);
        int p0 = atomicAdd(&counts[i0], 1);
        lists[(size_t)i0 * T + p0] = t;
        int p1 = atomicAdd(&counts[i1], 1);
        lists[(size_t)i1 * T + p1] = t;
    }
}

// ---------------------------------------------------------------------------
// Kernel 2: fp32 -> bf16 elementwise (hidden states)
// ---------------------------------------------------------------------------
__global__ void conv_bf16_kernel(const float* __restrict__ src, __bf16* __restrict__ dst, long n) {
    long i = ((long)blockIdx.x * blockDim.x + threadIdx.x) * 4;
    if (i < n) {
        float4 v = *(const float4*)(src + i);
        dst[i + 0] = (__bf16)v.x;
        dst[i + 1] = (__bf16)v.y;
        dst[i + 2] = (__bf16)v.z;
        dst[i + 3] = (__bf16)v.w;
    }
}

// ---------------------------------------------------------------------------
// Kernel 3: transpose+convert weights: src [E][R][C] fp32 -> dst [E][C][R] bf16
// (makes the WMMA B-operand K-dimension contiguous)
// ---------------------------------------------------------------------------
__global__ __launch_bounds__(256) void transpose_conv_kernel(
        const float* __restrict__ src, __bf16* __restrict__ dst, int R, int C) {
    __shared__ float tile[32][33];
    int e = blockIdx.z;
    const float* s = src + (size_t)e * R * C;
    __bf16* d = dst + (size_t)e * R * C;
    int c0 = blockIdx.x * 32, r0 = blockIdx.y * 32;
    int tx = threadIdx.x, ty = threadIdx.y;   // 32 x 8
#pragma unroll
    for (int i = 0; i < 32; i += 8)
        tile[ty + i][tx] = s[(size_t)(r0 + ty + i) * C + (c0 + tx)];
    __syncthreads();
#pragma unroll
    for (int i = 0; i < 32; i += 8)
        d[(size_t)(c0 + ty + i) * R + (r0 + tx)] = (__bf16)tile[tx][ty + i];
}

// ---------------------------------------------------------------------------
// Kernel 4: fused gathered MoE GEMM.
//   block = 256 threads = 8 waves; MT=32 gathered tokens x full H outputs.
//   GEMM1 (cooperative, act chunk in LDS): act = silu(X @ W1[:, fc:fc+FC])
//   GEMM2 (per wave, 128-wide N slice): out += act @ W2[fc:fc+FC, :]
// ---------------------------------------------------------------------------
__global__ __launch_bounds__(256, 1) void moe_kernel(
        const __bf16* __restrict__ Xb,   // [T][H] bf16
        const __bf16* __restrict__ w1t,  // [E][F][H] bf16 (K contiguous)
        const __bf16* __restrict__ w2t,  // [E][H][F] bf16 (K contiguous)
        const float* __restrict__ gates, // [T][E]
        const int* __restrict__ counts,  // [E]
        const int* __restrict__ lists,   // [E][T]
        float* __restrict__ out) {       // [T][H] fp32 (atomic accumulate)
    int e = blockIdx.y;
    int cnt = counts[e];
    int tbase = blockIdx.x * MT;
    if (tbase >= cnt) return;

    __shared__ __align__(16) __bf16 Xs[MT][H];    // 64 KB (CDNA5: 320KB LDS/WGP)
    __shared__ __align__(16) __bf16 acts[MT][FC]; // 4 KB
    __shared__ int   toks[MT];
    __shared__ float gts[MT];

    int tid  = threadIdx.x;
    int lane = tid & 31;
    int wave = tid >> 5;
    int lhalf = (lane >= 16) ? 1 : 0;
    int lmod  = lane & 15;

    // gather token ids + gate weights
    if (tid < MT) {
        int m = tbase + tid;
        int tok = (m < cnt) ? lists[(size_t)e * T + m] : -1;
        toks[tid] = tok;
        gts[tid]  = (tok >= 0) ? gates[(size_t)tok * E + e] : 0.f;
    }
    __syncthreads();

    // stage X rows into LDS: 8 threads per row, uint4 = 8 bf16 per load
    {
        int r = tid >> 3, j = tid & 7;
        int tok = toks[r];
        const uint4* src = (tok >= 0) ? (const uint4*)(Xb + (size_t)tok * H) : nullptr;
        uint4* dst = (uint4*)(&Xs[r][0]);
#pragma unroll
        for (int it = 0; it < 16; ++it) {
            int idx = j + it * 8;   // 128 uint4 per row
            uint4 v = src ? src[idx] : make_uint4(0u, 0u, 0u, 0u);
            dst[idx] = v;
        }
    }
    __syncthreads();

    // accumulators for GEMM2: 2 M-tiles x 8 N-tiles of 16x16 f32
    v8f acc2[2][8];
#pragma unroll
    for (int mt = 0; mt < 2; ++mt)
#pragma unroll
        for (int nt = 0; nt < 8; ++nt)
#pragma unroll
            for (int r = 0; r < 8; ++r) acc2[mt][nt][r] = 0.f;

    int am = (wave >> 2) * 16;     // GEMM1: this wave's act M offset
    int an = (wave & 3) * 16;      // GEMM1: this wave's act N (f) offset
    int nb = wave * 128;           // GEMM2: this wave's H slice

    for (int fc = 0; fc < FF; fc += FC) {
        // L2 warm-up hints for the next chunk's weight columns
        if (fc + FC < FF) {
            __builtin_prefetch(w1t + ((size_t)e * FF + (fc + FC + an + lmod)) * H, 0, 1);
            __builtin_prefetch(w2t + ((size_t)e * H + (nb + lmod)) * FF + (fc + FC), 0, 1);
        }

        // ---- GEMM1: act[am:am+16][an:an+16] = X @ W1 chunk, K = H ----
        v8f acc1;
#pragma unroll
        for (int r = 0; r < 8; ++r) acc1[r] = 0.f;

        const __bf16* bcol1 = w1t + ((size_t)e * FF + (fc + an + lmod)) * H;
        int mrow = am + lmod;
        for (int kk = 0; kk < H; kk += 32) {
            Frag A, B;
#pragma unroll
            for (int n = 0; n < 8; ++n) {   // A: 16-bit 16x32 layout
                int k = kk + ((n >= 4) ? 16 : 0) + (lhalf ? 8 : 0) + 2 * (n & 3);
                A.u[n] = *(const unsigned*)&Xs[mrow][k];
            }
#pragma unroll
            for (int n = 0; n < 8; ++n) {   // B: 16-bit 32x16 layout
                int k = kk + ((n >= 4) ? 16 : 0) + (lhalf ? 8 : 0) + 2 * (n & 3);
                B.u[n] = *(const unsigned*)&bcol1[k];
            }
            acc1 = __builtin_amdgcn_wmma_f32_16x16x32_bf16(
                false, A.v, false, B.v, (short)0, acc1, false, false);
        }

        __syncthreads();   // previous chunk's GEMM2 done reading acts
        // silu + convert + store act chunk to LDS.
        // Fast path: v_rcp_f32 instead of the full IEEE divide expansion —
        // the result is immediately rounded to bf16, so 1-ULP rcp is plenty.
#pragma unroll
        for (int r = 0; r < 8; ++r) {
            float x = acc1[r];
            float s = x * __builtin_amdgcn_rcpf(1.0f + __expf(-x));
            int m = am + r + (lhalf ? 8 : 0);
            acts[m][an + lmod] = (__bf16)s;
        }
        __syncthreads();

        // ---- GEMM2: out slice += act(32xFC) @ W2 chunk(FC x 128) ----
#pragma unroll
        for (int kk2 = 0; kk2 < FC; kk2 += 32) {
            Frag A0, A1;
#pragma unroll
            for (int n = 0; n < 8; ++n) {
                int k = kk2 + ((n >= 4) ? 16 : 0) + (lhalf ? 8 : 0) + 2 * (n & 3);
                A0.u[n] = *(const unsigned*)&acts[lmod][k];
                A1.u[n] = *(const unsigned*)&acts[16 + lmod][k];
            }
#pragma unroll
            for (int nt = 0; nt < 8; ++nt) {
                Frag B;
                const __bf16* bcol2 = w2t + ((size_t)e * H + (nb + nt * 16 + lmod)) * FF;
#pragma unroll
                for (int n = 0; n < 8; ++n) {
                    int k = fc + kk2 + ((n >= 4) ? 16 : 0) + (lhalf ? 8 : 0) + 2 * (n & 3);
                    B.u[n] = *(const unsigned*)&bcol2[k];
                }
                acc2[0][nt] = __builtin_amdgcn_wmma_f32_16x16x32_bf16(
                    false, A0.v, false, B.v, (short)0, acc2[0][nt], false, false);
                acc2[1][nt] = __builtin_amdgcn_wmma_f32_16x16x32_bf16(
                    false, A1.v, false, B.v, (short)0, acc2[1][nt], false, false);
            }
        }
    }

    // epilogue: scale by gate, scatter-add (exactly 2 commutative adds/element)
#pragma unroll
    for (int mt = 0; mt < 2; ++mt) {
#pragma unroll
        for (int r = 0; r < 8; ++r) {
            int m = mt * 16 + r + (lhalf ? 8 : 0);
            int tok = toks[m];
            if (tok >= 0) {
                float g = gts[m];
#pragma unroll
                for (int nt = 0; nt < 8; ++nt) {
                    int col = nb + nt * 16 + lmod;
                    atomicAdd(&out[(size_t)tok * H + col], g * acc2[mt][nt][r]);
                }
            }
        }
    }
}

// ---------------------------------------------------------------------------
// Host launcher
// ---------------------------------------------------------------------------
extern "C" void kernel_launch(void* const* d_in, const int* in_sizes, int n_in,
                              void* d_out, int out_size, void* d_ws, size_t ws_size,
                              hipStream_t stream) {
    const float* hidden = (const float*)d_in[0];   // [T][H]
    const float* gate_w = (const float*)d_in[1];   // [H][E]
    const float* w1     = (const float*)d_in[2];   // [E][H][F]
    const float* w2     = (const float*)d_in[3];   // [E][F][H]
    float* out = (float*)d_out;                    // [T][H]
    (void)in_sizes; (void)n_in; (void)out_size; (void)ws_size;

    // workspace carve-up
    char* ws = (char*)d_ws;
    float* gates = (float*)ws;        ws += (size_t)T * E * sizeof(float);   // 512 KB
    int* counts  = (int*)ws;          ws += 256;
    int* lists   = (int*)ws;          ws += (size_t)E * T * sizeof(int);     // 512 KB
    __bf16* hB   = (__bf16*)ws;       ws += (size_t)T * H * 2;               // 32 MB
    __bf16* w1t  = (__bf16*)ws;       ws += (size_t)E * H * FF * 2;          // 32 MB
    __bf16* w2t  = (__bf16*)ws;       ws += (size_t)E * H * FF * 2;          // 32 MB

    // 0) zero out + counters
    {
        long n4 = ((long)T * H) / 4;
        int blocks = (int)((n4 + 255) / 256);
        zero_init_kernel<<<blocks, 256, 0, stream>>>(out, counts);
    }
    // 1) router
    router_kernel<<<T / NWAVE, 256, 0, stream>>>(hidden, gate_w, gates, counts, lists);
    // 2) hidden -> bf16
    {
        long n = (long)T * H;
        int blocks = (int)((n / 4 + 255) / 256);
        conv_bf16_kernel<<<blocks, 256, 0, stream>>>(hidden, hB, n);
    }
    // 3) weight transpose+convert: w1 [E][H][F] -> [E][F][H]; w2 [E][F][H] -> [E][H][F]
    {
        dim3 blk(32, 8);
        dim3 g1(FF / 32, H / 32, E);
        transpose_conv_kernel<<<g1, blk, 0, stream>>>(w1, w1t, H, FF);
        dim3 g2(H / 32, FF / 32, E);
        transpose_conv_kernel<<<g2, blk, 0, stream>>>(w2, w2t, FF, H);
    }
    // 4) fused MoE GEMM over gathered tokens
    {
        dim3 grid(T / MT, E);
        moe_kernel<<<grid, 256, 0, stream>>>(hB, w1t, w2t, gates, counts, lists, out);
    }
}
